// Edge_embedding_tetris_inv_88656714925212
// MI455X (gfx1250) — matrix operand order repaired and checked by reading, not compiled, not run on hardware
//
#include <hip/hip_runtime.h>

typedef __attribute__((ext_vector_type(16))) _Float16 v16h;
typedef __attribute__((ext_vector_type(8)))  float    v8f;
typedef __attribute__((ext_vector_type(2)))  _Float16 h2;

#define HIDDEN 96
#define NSC    32

static __device__ __forceinline__ h2 mkh2(float a, float b) {
  h2 r; r.x = (_Float16)a; r.y = (_Float16)b; return r;
}

__global__ void __launch_bounds__(256) zero_out_kernel(float* __restrict__ out, int n) {
  int i = blockIdx.x * blockDim.x + threadIdx.x;
  if (i < n) out[i] = 0.0f;
}

// One wave processes 16 edges per tile:
//   layer1 (K=2) in packed f16 VALU, written directly into WMMA A layout,
//   layer2 (96x32) as 6x v_wmma_f32_16x16x32_f16,
//   scatter via cacheline-contiguous global_atomic_add_f32 into L2-resident out.
__global__ void __launch_bounds__(256) edge_mlp_scatter_kernel(
    const float* __restrict__ v,     // [E,2]
    const int*   __restrict__ col,   // [E] (second row of edge_index)
    const float* __restrict__ W1,    // [2,96]
    const float* __restrict__ b1,    // [96]
    const float* __restrict__ W2,    // [96,32]
    const float* __restrict__ b2,    // [32]
    float*       __restrict__ out,   // [N_nodes,32]
    int E) {
  const int lane  = threadIdx.x & 31;
  const int hhalf = lane >> 4;   // 0: lanes 0-15, 1: lanes 16-31
  const int nlo   = lane & 15;
  const unsigned nloB = (unsigned)nlo * 4u;   // per-lane byte offset in node row

  // ---- Preload W2 into per-lane registers in the WMMA B (32x16 f16) layout.
  // B block c covers K = 32c..32c+31; lane halves own K sub-blocks of 8:
  //   element j holds K = 32c + (j/8)*16 + hhalf*8 + j%8, column n = t*16 + nlo.
  v16h B[2][3];
#pragma unroll
  for (int t = 0; t < 2; ++t)
#pragma unroll
    for (int c = 0; c < 3; ++c)
#pragma unroll
      for (int j = 0; j < 16; ++j) {
        int K = 32 * c + (j >> 3) * 16 + hhalf * 8 + (j & 7);
        B[t][c][j] = (_Float16)W2[K * NSC + t * 16 + nlo];
      }

  // ---- Preload W1/b1 packed per-lane matching the WMMA A (16x32 f16) layout.
  // A register c, half-pair i (i=0..7) covers K,K+1 with
  //   K = 32c + (i/4)*16 + hhalf*8 + (i%4)*2
  h2 w0pk[3][8], w1pk[3][8], b1pk[3][8];
#pragma unroll
  for (int c = 0; c < 3; ++c)
#pragma unroll
    for (int i = 0; i < 8; ++i) {
      int K = 32 * c + (i >> 2) * 16 + hhalf * 8 + (i & 3) * 2;
      w0pk[c][i] = mkh2(W1[K],          W1[K + 1]);
      w1pk[c][i] = mkh2(W1[HIDDEN + K], W1[HIDDEN + K + 1]);
      b1pk[c][i] = mkh2(b1[K],          b1[K + 1]);
    }

  const float bias0 = b2[nlo];       // n = 0..15
  const float bias1 = b2[16 + nlo];  // n = 16..31
  const h2 zeroh2 = mkh2(0.0f, 0.0f);

  const int nWaves = (gridDim.x * blockDim.x) >> 5;
  const int waveId = (blockIdx.x * blockDim.x + threadIdx.x) >> 5;
  const int nFull  = E >> 4;          // full 16-edge tiles (E=3.2M -> all full)

  // Tile body. `guard` is compile-time-constant at both call sites; with
  // always-inline the tail checks fold away in the hot loop.
  auto do_tile = [&](int e0, bool guard) __attribute__((always_inline)) {
    const int e = e0 + nlo;           // lanes L and L+16 both own edge row L&15

    float v0 = 0.0f, v1 = 0.0f;
    unsigned colB = 0;                // byte offset of node row: col[e]*128
    if (!guard || e < E) {
      v0 = v[2 * e];
      v1 = v[2 * e + 1];
      colB = (unsigned)col[e] << 7;
    }
    const h2 hv0 = mkh2(v0, v0);
    const h2 hv1 = mkh2(v1, v1);

    // ---- Layer 1: h = relu(v0*W1[0] + v1*W1[1] + b1), built in A layout.
    v16h A[3];
#pragma unroll
    for (int c = 0; c < 3; ++c) {
#pragma unroll
      for (int i = 0; i < 8; ++i) {
        h2 t = hv1 * w1pk[c][i] + b1pk[c][i];     // v_pk_fma_f16
        t = hv0 * w0pk[c][i] + t;                 // v_pk_fma_f16
        t = __builtin_elementwise_max(t, zeroh2); // v_pk_max (ReLU)
        A[c][2 * i]     = t.x;
        A[c][2 * i + 1] = t.y;
      }
    }

    // ---- Layer 2: [16,96] x [96,32] via 6 WMMAs, f32 accumulate, bias in C.
    // Interleaved acc0/acc1 so adjacent WMMAs share A and avoid acc RAW.
    v8f acc0, acc1;
#pragma unroll
    for (int r = 0; r < 8; ++r) { acc0[r] = bias0; acc1[r] = bias1; }

    acc0 = __builtin_amdgcn_wmma_f32_16x16x32_f16(false, A[0], false, B[0][0], (short)0, acc0, false, false);
    acc1 = __builtin_amdgcn_wmma_f32_16x16x32_f16(false, A[0], false, B[1][0], (short)0, acc1, false, false);
    acc0 = __builtin_amdgcn_wmma_f32_16x16x32_f16(false, A[1], false, B[0][1], (short)0, acc0, false, false);
    acc1 = __builtin_amdgcn_wmma_f32_16x16x32_f16(false, A[1], false, B[1][1], (short)0, acc1, false, false);
    acc0 = __builtin_amdgcn_wmma_f32_16x16x32_f16(false, A[2], false, B[0][2], (short)0, acc0, false, false);
    acc1 = __builtin_amdgcn_wmma_f32_16x16x32_f16(false, A[2], false, B[1][2], (short)0, acc1, false, false);

    // ---- Scatter: D VGPR r, lane L -> edge m = r + 8*(L/16), column n = L&15.
    // Lanes 0-15 of a given r hit 16 consecutive floats of one node row ->
    // one-cacheline f32 atomics (out fits in the 192MB L2). 32-bit byte
    // offsets keep the saddr+voffset atomic form: 1 v_add per r.
#pragma unroll
    for (int r = 0; r < 8; ++r) {
      const int m = r + 8 * hhalf;
      const unsigned rowB = (unsigned)__shfl((int)colB, m, 32); // ds_bpermute
      if (!guard || (e0 + m < E)) {
        float* p = (float*)((char*)out + (rowB + nloB));
        unsafeAtomicAdd(p,      acc0[r]);   // bytes [0,64)   of node row
        unsafeAtomicAdd(p + 16, acc1[r]);   // bytes [64,128) of node row
      }
    }
  };

  // Hot loop: full tiles, no bounds checks anywhere.
  for (int tile = waveId; tile < nFull; tile += nWaves)
    do_tile(tile << 4, false);

  // At most one partial tail tile; wave 0 only.
  if ((E & 15) && waveId == 0)
    do_tile(nFull << 4, true);
}

extern "C" void kernel_launch(void* const* d_in, const int* in_sizes, int n_in,
                              void* d_out, int out_size, void* d_ws, size_t ws_size,
                              hipStream_t stream) {
  const float* v    = (const float*)d_in[0];
  const int*   eidx = (const int*)d_in[1];   // [2,E] flattened
  const float* W1   = (const float*)d_in[2];
  const float* b1   = (const float*)d_in[3];
  const float* W2   = (const float*)d_in[4];
  const float* b2   = (const float*)d_in[5];
  float* out = (float*)d_out;

  const int E = in_sizes[0] / 2;         // v is [E,2]
  const int* col = eidx + E;             // edge_index[1]

  // Zero the (harness-poisoned) output accumulator first, same stream.
  zero_out_kernel<<<(out_size + 255) / 256, 256, 0, stream>>>(out, out_size);

  const int nTiles = (E + 15) >> 4;
  // ~16 tiles per block (8 waves): amortizes the one-time per-wave weight
  // preload (~170 loads/lane) while keeping ~16K waves of parallelism.
  int blocks = (nTiles + 15) / 16;
  if (blocks > 2048) blocks = 2048;
  if (blocks < 1) blocks = 1;
  edge_mlp_scatter_kernel<<<blocks, 256, 0, stream>>>(v, col, W1, b1, W2, b2, out, E);
}
